// NTXentLoss_55027120996804
// MI455X (gfx1250) — matrix operand order repaired and checked by reading, not compiled, not run on hardware
//
#include <hip/hip_runtime.h>
#include <hip/hip_bf16.h>

typedef __bf16          v16bf  __attribute__((ext_vector_type(16)));
typedef float           v8f    __attribute__((ext_vector_type(8)));
typedef unsigned short  u16x8  __attribute__((ext_vector_type(8)));
typedef float           f32x4  __attribute__((ext_vector_type(4)));
typedef unsigned int    u32x4  __attribute__((ext_vector_type(4)));
typedef int             i32x4  __attribute__((ext_vector_type(4)));
typedef int             i32x8  __attribute__((ext_vector_type(8)));

#define INV_TAU     (1.0f / 0.07f)
#define POS_THRESH  0.2f
#define Bsz         4096
#define Dd          512
#define TWOB        8192

// LDS panel: 16 columns x 512 bf16, TDM-padded by 16B per 1024B row
// -> row stride 520 elements (breaks the 64-bank conflict pattern).
#define PANEL_STRIDE_E  520
#define PANEL_E         (16 * PANEL_STRIDE_E)

// ---------------------------------------------------------------------------
// Normalize rows of [emb_i; emb_j] (fp32) -> bf16 matrix Nmat (TWOB x Dd).
// ---------------------------------------------------------------------------
__global__ void nx_normalize(const float* __restrict__ emb_i,
                             const float* __restrict__ emb_j,
                             __bf16* __restrict__ Nmat) {
    const int wid  = threadIdx.x >> 5;
    const int lane = threadIdx.x & 31;
    const int row  = blockIdx.x * 4 + wid;
    if (row >= TWOB) return;

    const float* src = (row < Bsz) ? (emb_i + (size_t)row * Dd)
                                   : (emb_j + (size_t)(row - Bsz) * Dd);
    float v[16];
    const f32x4* s4 = (const f32x4*)(src + lane * 16);
    *(f32x4*)(v + 0)  = s4[0];
    *(f32x4*)(v + 4)  = s4[1];
    *(f32x4*)(v + 8)  = s4[2];
    *(f32x4*)(v + 12) = s4[3];

    float ss = 0.f;
#pragma unroll
    for (int i = 0; i < 16; ++i) ss += v[i] * v[i];
#pragma unroll
    for (int m = 1; m < 32; m <<= 1) ss += __shfl_xor(ss, m, 32);
    const float r = rsqrtf(ss);

    union { __bf16 b[16]; u16x8 u[2]; } o;
#pragma unroll
    for (int i = 0; i < 16; ++i) o.b[i] = (__bf16)(v[i] * r);

    u16x8* dst = (u16x8*)((unsigned short*)Nmat + (size_t)row * Dd + lane * 16);
    dst[0] = o.u[0];
    dst[1] = o.u[1];
}

// ---------------------------------------------------------------------------
// pairdot[k] = dot(Nmat[k], Nmat[B+k]) in f32.
// ---------------------------------------------------------------------------
__global__ void nx_pairdot(const __bf16* __restrict__ Nmat,
                           float* __restrict__ pairdot) {
    const int wid  = threadIdx.x >> 5;
    const int lane = threadIdx.x & 31;
    const int k    = blockIdx.x * 4 + wid;
    if (k >= Bsz) return;

    const __bf16* a = Nmat + (size_t)k * Dd + lane * 16;
    const __bf16* b = Nmat + (size_t)(k + Bsz) * Dd + lane * 16;
    float s = 0.f;
#pragma unroll
    for (int i = 0; i < 16; ++i) s += (float)a[i] * (float)b[i];
#pragma unroll
    for (int m = 1; m < 32; m <<= 1) s += __shfl_xor(s, m, 32);
    if (lane == 0) pairdot[k] = s;
}

// ---------------------------------------------------------------------------
// TDM: async-load one 16-column x 512-elem bf16 panel (tile) into LDS,
// with hardware padding of 4 dwords after every 256 dwords (1024B row).
// D# layout per CDNA5 ISA.
// This toolchain exposes the 6-arg builtin:
//   (u32x4 g0, i32x8 g1, i32x4 g2, i32x4 g3, i32x8 extra, i32 cpol)
// groups 2/3 and the trailing group are unused (zero) for a 2-D tile.
// ---------------------------------------------------------------------------
__device__ __forceinline__ void tdm_load_panel(unsigned lds_addr,
                                               unsigned long long gaddr) {
    u32x4 g0;
    g0[0] = 1u;                                       // count=1, user mode
    g0[1] = lds_addr;                                 // LDS byte address
    g0[2] = (unsigned)(gaddr & 0xFFFFFFFFull);        // global_addr[31:0]
    g0[3] = (unsigned)((gaddr >> 32) & 0x1FFFFFFull)  // global_addr[56:32]
          | (2u << 30);                               // type = 2 ("image")

    i32x8 g1;
    g1[0] = (int)((1u << 16)      // data_size = 1 -> 2 bytes
                | (1u << 20)      // pad_enable
                | (7u << 22)      // pad_interval: 256 dwords (=1024B row)
                | (3u << 25));    // pad_amount:   4 dwords  (=16B)
    g1[1] = (int)((Dd & 0xFFFFu) << 16);      // tensor_dim0 = 512 (lo16)
    g1[2] = (int)((TWOB & 0xFFFFu) << 16);    // dim0 hi=0 | tensor_dim1 lo16
    g1[3] = (int)((Dd & 0xFFFFu) << 16);      // dim1 hi=0 | tile_dim0 = 512
    g1[4] = 16;                               // tile_dim1 = 16 | tile_dim2 = 0
    g1[5] = Dd;                               // tensor_dim0_stride lo32 = 512
    g1[6] = 0;                                // stride0 hi | stride1 lo
    g1[7] = 0;

    i32x4 g2 = {0, 0, 0, 0};                  // 2-D tile: groups 2/3 unused
    i32x4 g3 = {0, 0, 0, 0};
    i32x8 gx = {0, 0, 0, 0, 0, 0, 0, 0};      // trailing group: unused

    __builtin_amdgcn_tensor_load_to_lds(g0, g1, g2, g3, gx, 0);
}

// ---------------------------------------------------------------------------
// Hot kernel: rowsum[r] += sum_{c != r} exp((dot_rc - 1)/tau).
// 8 waves x 16 rows = 128 rows/block; A cached in 128 VGPRs/wave.
// B panels (16 cols x 512) TDM double-buffered in LDS, read via ds_load_b128.
// ---------------------------------------------------------------------------
union FragU { v16bf v; struct { u16x8 lo, hi; } h; };

__global__ void __launch_bounds__(256)
nx_gemm_expsum(const __bf16* __restrict__ Nmat, float* __restrict__ rowsum) {
    __shared__ __align__(16) unsigned short sB[2][PANEL_E];

    const int lane    = threadIdx.x & 31;
    const int wid     = threadIdx.x >> 5;
    const int half    = lane >> 4;
    const int l16     = lane & 15;
    const int rowBase = blockIdx.x * 128 + wid * 16;
    const int colBase = blockIdx.y * (TWOB / 8);     // 1024-column chunk
    const int nTiles  = (TWOB / 8) / 16;             // 64 tiles of 16 cols

    // Cache A fragments: wave's 16 rows, K=512 (contiguous k order, same as B:
    // elems 0..15 = K[k0+half*16, +16) contiguous).
    const unsigned short* rp =
        (const unsigned short*)Nmat + (size_t)(rowBase + l16) * Dd;
    FragU a[16];
#pragma unroll
    for (int ks = 0; ks < 16; ++ks) {
        const int k0 = ks * 32 + half * 16;
        a[ks].h.lo = *(const u16x8*)(rp + k0);
        a[ks].h.hi = *(const u16x8*)(rp + k0 + 8);
    }

    // Generic->LDS: low 32 bits of a flat shared pointer are the LDS offset.
    const unsigned ldsBase = (unsigned)(size_t)(void*)&sB[0][0];
    const unsigned long long gBase =
        (unsigned long long)(size_t)Nmat + (size_t)colBase * Dd * 2;

    // Prologue: wave 0 pulls tile 0 into buffer 0.
    if (wid == 0) {
        tdm_load_panel(ldsBase, gBase);
        __builtin_amdgcn_s_wait_tensorcnt(0);
    }
    __syncthreads();

    float racc[8];
#pragma unroll
    for (int v = 0; v < 8; ++v) racc[v] = 0.f;
    const int myRow0 = rowBase + half * 8;

    const unsigned short* myCol = &sB[0][0] + l16 * PANEL_STRIDE_E;

    for (int ct = 0; ct < nTiles; ++ct) {
        const int buf = ct & 1;
        // Prefetch next panel into the other buffer (safe: that buffer was
        // last read in iteration ct-1, fenced by the trailing barrier).
        if (wid == 0 && ct + 1 < nTiles) {
            tdm_load_panel(ldsBase + (unsigned)(buf ^ 1) * PANEL_E * 2,
                           gBase + (unsigned long long)(ct + 1) * 16 * Dd * 2);
        }

        const unsigned short* cp = myCol + buf * PANEL_E;
        const int col = colBase + ct * 16 + l16;

        // B (16-bit 32x16), contiguous k order as A: lane = column l16, elems 0..15 =
        // K[k0 + half*16, +16) contiguous. Register-double-buffer the ds
        // loads so step k+1 is in flight during the WMMA of step k.
        v8f c = {};
        FragU b, nb;
        b.h.lo = *(const u16x8*)(cp + half * 16);
        b.h.hi = *(const u16x8*)(cp + half * 16 + 8);
#pragma unroll
        for (int ks = 0; ks < 16; ++ks) {
            const int kn = ((ks + 1) & 15) * 32 + half * 16;
            nb.h.lo = *(const u16x8*)(cp + kn);
            nb.h.hi = *(const u16x8*)(cp + kn + 8);
            c = __builtin_amdgcn_wmma_f32_16x16x32_bf16(
                    false, a[ks].v, false, b.v, (short)0, c, false, false);
            b = nb;
        }

        // C layout: lane's element v holds (M = half*8 + v, N = col).
        // cos <= 1 => exp((dot-1)/tau) <= 1: constant shift, no online max.
#pragma unroll
        for (int v = 0; v < 8; ++v) {
            float t = __expf((c[v] - 1.0f) * INV_TAU);
            if (myRow0 + v == col) t = 0.f;   // mask self-similarity
            racc[v] += t;
        }

        if (wid == 0) __builtin_amdgcn_s_wait_tensorcnt(0);
        __syncthreads();
    }

    // Reduce row partials across the 16 lanes of each half, one atomic/row.
#pragma unroll
    for (int v = 0; v < 8; ++v) {
        float s = racc[v];
#pragma unroll
        for (int m = 1; m < 16; m <<= 1) s += __shfl_xor(s, m, 32);
        if (l16 == 0) atomicAdd(&rowsum[myRow0 + v], s);
    }
}

// ---------------------------------------------------------------------------
__global__ void nx_zero(float* __restrict__ p, int n) {
    int i = blockIdx.x * blockDim.x + threadIdx.x;
    if (i < n) p[i] = 0.f;
}

__global__ void nx_finalize(const float* __restrict__ rowsum,
                            const float* __restrict__ pairdot,
                            const float* __restrict__ dist,
                            float* __restrict__ out) {
    __shared__ float sl[256];
    __shared__ float sc[256];
    float ls = 0.f, cnt = 0.f;
    for (int k = threadIdx.x; k < Bsz; k += 256) {
        if (dist[k] < POS_THRESH) {
            float lp = 2.0f * (pairdot[k] - 1.0f) * INV_TAU
                     - logf(rowsum[k]) - logf(rowsum[k + Bsz]);
            ls += -lp;
            cnt += 2.0f;
        }
    }
    sl[threadIdx.x] = ls;
    sc[threadIdx.x] = cnt;
    __syncthreads();
    for (int s = 128; s > 0; s >>= 1) {
        if (threadIdx.x < (unsigned)s) {
            sl[threadIdx.x] += sl[threadIdx.x + s];
            sc[threadIdx.x] += sc[threadIdx.x + s];
        }
        __syncthreads();
    }
    if (threadIdx.x == 0) {
        float n = sc[0];
        out[0] = (n > 0.f) ? sl[0] / fmaxf(n, 1.0f) : 0.f;
    }
}

// ---------------------------------------------------------------------------
extern "C" void kernel_launch(void* const* d_in, const int* in_sizes, int n_in,
                              void* d_out, int out_size, void* d_ws, size_t ws_size,
                              hipStream_t stream) {
    (void)in_sizes; (void)n_in; (void)out_size; (void)ws_size;
    const float* emb_i = (const float*)d_in[0];
    const float* emb_j = (const float*)d_in[1];
    const float* dist  = (const float*)d_in[2];
    float* out = (float*)d_out;

    char* ws = (char*)d_ws;
    __bf16* Nmat   = (__bf16*)ws;                                   // 8 MB
    float*  rowsum = (float*)(ws + (size_t)TWOB * Dd * 2);          // 32 KB
    float*  pairs  = (float*)(ws + (size_t)TWOB * Dd * 2 + TWOB * 4);

    nx_zero<<<TWOB / 256, 256, 0, stream>>>(rowsum, TWOB);
    nx_normalize<<<TWOB / 4, 128, 0, stream>>>(emb_i, emb_j, Nmat);
    nx_pairdot<<<Bsz / 4, 128, 0, stream>>>(Nmat, pairs);
    nx_gemm_expsum<<<dim3(TWOB / 128, 8), 256, 0, stream>>>(Nmat, rowsum);
    nx_finalize<<<1, 256, 0, stream>>>(rowsum, pairs, dist, out);
}
